// GTransCYPs_78761110274541
// MI455X (gfx1250) — matrix operand order repaired and checked by reading, not compiled, not run on hardware
//
#include <hip/hip_runtime.h>
#include <hip/hip_bf16.h>

// ---------------------------------------------------------------------------
// Problem constants (match reference)
// ---------------------------------------------------------------------------
#define NN_NODES 100000
#define NE_EDGES 400000
#define F_IN     128
#define EMB      64
#define HEADS    4
#define HC       256
#define EDGE_DIM 16
#define N_GRAPHS 2048
#define DENSE    256
#define BN_EPS   1e-5f

// ---------------------------------------------------------------------------
// WMMA types (probe-verified for gfx1250)
// ---------------------------------------------------------------------------
typedef __attribute__((ext_vector_type(16))) __bf16 v16bf;
typedef __attribute__((ext_vector_type(8)))  __bf16 v8bf;
typedef __attribute__((ext_vector_type(8)))  float  v8f;

// ---------------------------------------------------------------------------
// Ordered-bits mapping: unsigned compare order == float compare order.
// Lets us do float segment-max with native global_atomic_max_u32.
// ---------------------------------------------------------------------------
__device__ __forceinline__ unsigned f2ord(float f) {
    unsigned u = __float_as_uint(f);
    return (u & 0x80000000u) ? ~u : (u | 0x80000000u);
}
__device__ __forceinline__ float ord2f(unsigned u) {
    return __uint_as_float((u & 0x80000000u) ? (u & 0x7fffffffu) : ~u);
}

// pack two f32 into one dword of bf16 pair
__device__ __forceinline__ unsigned pack_bf16(float a, float b) {
    union { __bf16 h[2]; unsigned u; } un;
    un.h[0] = (__bf16)a;
    un.h[1] = (__bf16)b;
    return un.u;
}

// ---------------------------------------------------------------------------
// Generic bf16-WMMA GEMM:  C[M,N] = epilogue( A[M,K] @ B[K,N] )
// mode: 0 = none, 1 = +bias, 2 = +bias,ReLU, 3 = +bias,ReLU,BatchNorm
// BM=128 BN=64 BK=32, 256 threads = 8 wave32, each wave: 16 rows x 64 cols
// => 4 v_wmma_f32_16x16x32_bf16 per wave per K-step (32 per block).
// A tile is row-major bf16 (row stride 40 elem = 80 B, multiple of 16 B);
// B tile is stored TRANSPOSED (Bt[col][k], stride 40 elem) so every WMMA
// fragment is two contiguous, 16B-aligned chunks -> 2x ds_load_b128 each.
// ---------------------------------------------------------------------------
#define BM 128
#define BN 64
#define BK 32
#define LDA 40   // A LDS row stride (elements); 80 B, multiple of 16 B
#define LDT 40   // Bt LDS row stride (elements); 80 B, multiple of 16 B

__global__ __launch_bounds__(256) void wmma_gemm_kernel(
    const float* __restrict__ A, const float* __restrict__ B,
    const float* __restrict__ bias, float* __restrict__ C,
    int M, int K, int N, int mode,
    const float* __restrict__ bn_g, const float* __restrict__ bn_b,
    const float* __restrict__ bn_m, const float* __restrict__ bn_v)
{
    __shared__ __bf16 As[BM * LDA];      // 10240 B
    __shared__ __bf16 Bt[BN * LDT];      // 5120 B (transposed: [col][k])

    const int tid  = threadIdx.x;
    const int wave = tid >> 5;
    const int lane = tid & 31;
    const int m0   = blockIdx.x * BM;
    const int n0   = blockIdx.y * BN;

    v8f acc[4];
#pragma unroll
    for (int t = 0; t < 4; ++t)
#pragma unroll
        for (int e = 0; e < 8; ++e) acc[t][e] = 0.0f;

    for (int k0 = 0; k0 < K; k0 += BK) {
        // ---- stage A tile (BM x BK) as bf16 pairs: b64 global, b32 LDS ----
        for (int i = tid; i < BM * BK / 2; i += 256) {
            int row = i >> 4;                 // 0..127
            int kp  = (i & 15) * 2;           // even k within tile
            int gr  = m0 + row, gk = k0 + kp;
            unsigned packed = 0u;
            if (gr < M && gk < K) {           // K even -> pair fully in/out
                const float2 av = *(const float2*)&A[(size_t)gr * K + gk];
                packed = pack_bf16(av.x, av.y);
            }
            *(unsigned*)&As[row * LDA + kp] = packed;
        }
        // ---- stage B tile (BK x BN) transposed into Bt[col][k] ----
        for (int i = tid; i < BK * BN / 2; i += 256) {
            int krow = i >> 5;                // 0..31
            int cp   = (i & 31) * 2;          // even col within tile
            int gk   = k0 + krow;
            float bx = 0.0f, by = 0.0f;
            if (gk < K) {
                const float2 bv = *(const float2*)&B[(size_t)gk * N + n0 + cp];
                bx = bv.x; by = bv.y;
            }
            Bt[(cp + 0) * LDT + krow] = (__bf16)bx;
            Bt[(cp + 1) * LDT + krow] = (__bf16)by;
        }
        __syncthreads();

        // ---- A fragment: 16x32 bf16 per ISA layout ----
        // elements 0-7  <-> K = akb .. akb+7      (contiguous 16 B)
        // elements 8-15 <-> K = akb+16 .. akb+23  (contiguous 16 B)
        const int arow = wave * 16 + (lane & 15);
        const int akb  = (lane >> 4) * 8;
        const v8bf alo = *(const v8bf*)&As[arow * LDA + akb];
        const v8bf ahi = *(const v8bf*)&As[arow * LDA + akb + 16];
        v16bf afrag = __builtin_shufflevector(alo, ahi,
            0, 1, 2, 3, 4, 5, 6, 7, 8, 9, 10, 11, 12, 13, 14, 15);

#pragma unroll
        for (int nt = 0; nt < 4; ++nt) {
            // ---- B fragment: elements 0-7 <-> K = bkb..bkb+7, 8-15 <-> K = bkb+16..bkb+23 ----
            const int bcol = nt * 16 + (lane & 15);
            const int bkb  = (lane >> 4) * 8;
            const v8bf blo = *(const v8bf*)&Bt[bcol * LDT + bkb];
            const v8bf bhi = *(const v8bf*)&Bt[bcol * LDT + bkb + 16];
            v16bf bfrag = __builtin_shufflevector(blo, bhi,
                0, 1, 2, 3, 4, 5, 6, 7, 8, 9, 10, 11, 12, 13, 14, 15);

            acc[nt] = __builtin_amdgcn_wmma_f32_16x16x32_bf16(
                false, afrag, false, bfrag, (short)0, acc[nt], false, false);
        }
        __syncthreads();
    }

    // ---- epilogue: C layout, VGPR r -> M=r (lanes 0-15) / M=r+8 (16-31) ----
    const int rbase = m0 + wave * 16 + ((lane >> 4) ? 8 : 0);
#pragma unroll
    for (int nt = 0; nt < 4; ++nt) {
        const int col = n0 + nt * 16 + (lane & 15);
#pragma unroll
        for (int rr = 0; rr < 8; ++rr) {
            int row = rbase + rr;
            if (row < M) {
                float vv = acc[nt][rr];
                if (mode >= 1) vv += bias[col];
                if (mode >= 2) vv = fmaxf(vv, 0.0f);
                if (mode >= 3)
                    vv = (vv - bn_m[col]) * rsqrtf(bn_v[col] + BN_EPS) * bn_g[col] + bn_b[col];
                C[(size_t)row * N + col] = vv;
            }
        }
    }
}

// ---------------------------------------------------------------------------
// Edge pass 1: alpha[e,h] = <q[dst], k[src]+e_proj> / 8 ; segment max into amax
// One thread per (edge, head), float4 streaming loads (global_load_b128).
// ---------------------------------------------------------------------------
__global__ void edge_alpha_kernel(const float* __restrict__ q,
                                  const float* __restrict__ k,
                                  const float* __restrict__ eproj,
                                  const int* __restrict__ src,
                                  const int* __restrict__ dst,
                                  float* __restrict__ alpha,
                                  unsigned* __restrict__ amaxU)
{
    int idx = blockIdx.x * blockDim.x + threadIdx.x;
    if (idx >= NE_EDGES * HEADS) return;
    int e = idx >> 2, hh = idx & 3;
    int s = src[e], d = dst[e];
    const float4* q4 = (const float4*)(q + (size_t)d * HC + hh * EMB);
    const float4* k4 = (const float4*)(k + (size_t)s * HC + hh * EMB);
    const float4* e4 = (const float4*)(eproj + (size_t)e * HC + hh * EMB);
    float a = 0.0f;
#pragma unroll
    for (int c = 0; c < EMB / 4; ++c) {
        float4 qv = q4[c], kv = k4[c], ev = e4[c];
        a += qv.x * (kv.x + ev.x) + qv.y * (kv.y + ev.y)
           + qv.z * (kv.z + ev.z) + qv.w * (kv.w + ev.w);
    }
    a *= 0.125f;   // 1/sqrt(EMB)
    alpha[idx] = a;
    atomicMax(&amaxU[d * HEADS + hh], f2ord(a));
}

// ---------------------------------------------------------------------------
// Edge pass 2: ex = exp(alpha - amax[dst]) ; segment-sum into denom
// ---------------------------------------------------------------------------
__global__ void edge_exp_kernel(float* __restrict__ alpha,
                                const unsigned* __restrict__ amaxU,
                                float* __restrict__ denom,
                                const int* __restrict__ dst)
{
    int idx = blockIdx.x * blockDim.x + threadIdx.x;
    if (idx >= NE_EDGES * HEADS) return;
    int e = idx >> 2, hh = idx & 3;
    int d = dst[e];
    float m  = ord2f(amaxU[d * HEADS + hh]);
    float ex = __expf(alpha[idx] - m);
    alpha[idx] = ex;
    atomicAdd(&denom[d * HEADS + hh], ex);
}

// ---------------------------------------------------------------------------
// Edge pass 3: scatter messages: out[dst] += (v[src]+e_proj) * ex/denom
// One thread per (edge, channel).
// ---------------------------------------------------------------------------
__global__ void edge_scatter_kernel(const float* __restrict__ v,
                                    const float* __restrict__ eproj,
                                    const float* __restrict__ alpha,
                                    const float* __restrict__ denom,
                                    const int* __restrict__ src,
                                    const int* __restrict__ dst,
                                    float* __restrict__ outb)
{
    long long i = (long long)blockIdx.x * blockDim.x + threadIdx.x;
    if (i >= (long long)NE_EDGES * HC) return;
    int e = (int)(i >> 8);
    int c = (int)(i & 255);
    int hh = c >> 6;
    int s = src[e], d = dst[e];
    float w = alpha[e * HEADS + hh] / (denom[d * HEADS + hh] + 1e-16f);
    float msg = (v[(size_t)s * HC + c] + eproj[(size_t)e * HC + c]) * w;
    atomicAdd(&outb[(size_t)d * HC + c], msg);
}

// ---------------------------------------------------------------------------
// Beta gate: beta = sigmoid([out, r, out-r] @ Wbeta); out = beta*r+(1-beta)*out
// One wave32 per node; shuffle reduction for the 768-dot.
// ---------------------------------------------------------------------------
__global__ void beta_combine_kernel(float* __restrict__ outb,
                                    const float* __restrict__ r,
                                    const float* __restrict__ Wb)
{
    int gtid = blockIdx.x * blockDim.x + threadIdx.x;
    int node = gtid >> 5, lane = gtid & 31;
    if (node >= NN_NODES) return;
    const float* o  = outb + (size_t)node * HC;
    const float* rr = r    + (size_t)node * HC;
    float part = 0.0f;
    for (int c = lane; c < HC; c += 32) {
        float ov = o[c], rv = rr[c];
        part += ov * Wb[c] + rv * Wb[HC + c] + (ov - rv) * Wb[2 * HC + c];
    }
    for (int off = 16; off > 0; off >>= 1) part += __shfl_down(part, off, 32);
    float beta = 1.0f / (1.0f + __expf(-__shfl(part, 0, 32)));
    for (int c = lane; c < HC; c += 32) {
        float ov = o[c], rv = rr[c];
        outb[(size_t)node * HC + c] = beta * rv + (1.0f - beta) * ov;
    }
}

// ---------------------------------------------------------------------------
// hsum += h
// ---------------------------------------------------------------------------
__global__ void add_into_kernel(float* __restrict__ dstv,
                                const float* __restrict__ srcv, int n)
{
    int i = blockIdx.x * blockDim.x + threadIdx.x;
    if (i < n) dstv[i] += srcv[i];
}

// ---------------------------------------------------------------------------
// Gate projection (64-dot per node, wave32/node) + per-graph max
// ---------------------------------------------------------------------------
__global__ void gate_kernel(const float* __restrict__ hsum,
                            const float* __restrict__ gW,
                            const float* __restrict__ gb,
                            const int* __restrict__ batch,
                            float* __restrict__ gate,
                            unsigned* __restrict__ gmaxU)
{
    int gtid = blockIdx.x * blockDim.x + threadIdx.x;
    int node = gtid >> 5, lane = gtid & 31;
    if (node >= NN_NODES) return;
    float part = 0.0f;
    for (int c = lane; c < EMB; c += 32)
        part += hsum[(size_t)node * EMB + c] * gW[c];
    for (int off = 16; off > 0; off >>= 1) part += __shfl_down(part, off, 32);
    if (lane == 0) {
        float g = part + gb[0];
        gate[node] = g;
        atomicMax(&gmaxU[batch[node]], f2ord(g));
    }
}

__global__ void gate_exp_kernel(float* __restrict__ gate,
                                const unsigned* __restrict__ gmaxU,
                                float* __restrict__ gden,
                                const int* __restrict__ batch)
{
    int n = blockIdx.x * blockDim.x + threadIdx.x;
    if (n >= NN_NODES) return;
    int b = batch[n];
    float ge = __expf(gate[n] - ord2f(gmaxU[b]));
    gate[n] = ge;
    atomicAdd(&gden[b], ge);
}

__global__ void pool_scatter_kernel(const float* __restrict__ gate,
                                    const float* __restrict__ gden,
                                    const float* __restrict__ hsum,
                                    const int* __restrict__ batch,
                                    float* __restrict__ pooled)
{
    int i = blockIdx.x * blockDim.x + threadIdx.x;
    if (i >= NN_NODES * EMB) return;
    int n = i >> 6, c = i & 63;
    int b = batch[n];
    float att = gate[n] / (gden[b] + 1e-16f);
    atomicAdd(&pooled[b * EMB + c], att * hsum[i]);
}

// ---------------------------------------------------------------------------
// Final lin3: out[g] = z2[g,:] @ W[128] + b
// ---------------------------------------------------------------------------
__global__ void final_dot_kernel(const float* __restrict__ z2,
                                 const float* __restrict__ W,
                                 const float* __restrict__ b,
                                 float* __restrict__ out)
{
    int g = blockIdx.x * blockDim.x + threadIdx.x;
    if (g >= N_GRAPHS) return;
    float a = 0.0f;
#pragma unroll 4
    for (int j = 0; j < DENSE / 2; ++j) a += z2[g * (DENSE / 2) + j] * W[j];
    out[g] = a + b[0];
}

// ---------------------------------------------------------------------------
// Host-side orchestration
// ---------------------------------------------------------------------------
struct ConvP { const float *Wq,*bq,*Wk,*bk,*Wv,*bv,*We,*Ws,*bs,*Wbeta; };
struct BnP   { const float *g,*b,*m,*v; };

extern "C" void kernel_launch(void* const* d_in, const int* in_sizes, int n_in,
                              void* d_out, int out_size, void* d_ws, size_t ws_size,
                              hipStream_t stream) {
    (void)in_sizes; (void)n_in; (void)out_size; (void)ws_size;

    const float* x         = (const float*)d_in[0];
    const float* edge_attr = (const float*)d_in[1];
    const int*   eidx      = (const int*)d_in[2];
    const int*   batch     = (const int*)d_in[3];
    const int*   src = eidx;
    const int*   dst = eidx + NE_EDGES;

    // ---- params: flattened depth-first in dict insertion order ----
    int pi = 4;
    auto P = [&](void) { return (const float*)d_in[pi++]; };
    ConvP cv[4]; const float* tfW[4]; const float* tfB[4]; BnP bnp[4];
    for (int L = 0; L < 4; ++L) {
        cv[L].Wq = P(); cv[L].bq = P();
        cv[L].Wk = P(); cv[L].bk = P();
        cv[L].Wv = P(); cv[L].bv = P();
        cv[L].We = P();
        cv[L].Ws = P(); cv[L].bs = P();
        cv[L].Wbeta = P();
        tfW[L] = P(); tfB[L] = P();
        bnp[L].g = P(); bnp[L].b = P(); bnp[L].m = P(); bnp[L].v = P();
    }
    const float* gateW = P(); const float* gateb = P();
    const float* l1W = P();   const float* l1b = P();
    const float* l2W = P();   const float* l2b = P();
    const float* l3W = P();   const float* l3b = P();

    // ---- workspace layout (floats) ----
    float* ws = (float*)d_ws;
    size_t off = 0;
    auto alloc = [&](size_t n) { float* p = ws + off; off += n; return p; };
    const size_t NQ = (size_t)NN_NODES * HC;          // 25.6M
    float* q     = alloc(NQ);
    float* k     = alloc(NQ);
    float* v     = alloc(NQ);
    float* r     = alloc(NQ);
    float* outb  = alloc(NQ);
    float* eproj = alloc((size_t)NE_EDGES * HC);      // 102.4M
    float* alpha = alloc((size_t)NE_EDGES * HEADS);
    unsigned* amaxU = (unsigned*)alloc((size_t)NN_NODES * HEADS);
    float* denom = alloc((size_t)NN_NODES * HEADS);
    float* h     = alloc((size_t)NN_NODES * EMB);
    float* hsum  = alloc((size_t)NN_NODES * EMB);
    float* gate  = alloc(NN_NODES);
    unsigned* gmaxU = (unsigned*)alloc(N_GRAPHS);
    float* gden  = alloc(N_GRAPHS);
    float* pooled = alloc((size_t)N_GRAPHS * EMB);
    float* z1    = alloc((size_t)N_GRAPHS * DENSE);
    float* z2    = alloc((size_t)N_GRAPHS * (DENSE / 2));

    auto gemm = [&](const float* A, const float* B, const float* bias, float* C,
                    int M, int K, int N, int mode, const BnP* bn) {
        dim3 grid((M + BM - 1) / BM, N / BN);
        wmma_gemm_kernel<<<grid, 256, 0, stream>>>(
            A, B, bias, C, M, K, N, mode,
            bn ? bn->g : nullptr, bn ? bn->b : nullptr,
            bn ? bn->m : nullptr, bn ? bn->v : nullptr);
    };

    hipMemsetAsync(hsum, 0, (size_t)NN_NODES * EMB * sizeof(float), stream);

    const float* hin = x;
    int din = F_IN;
    for (int L = 0; L < 4; ++L) {
        // node projections (bf16 WMMA, f32 accumulate)
        gemm(hin, cv[L].Wq, cv[L].bq, q, NN_NODES, din, HC, 1, nullptr);
        gemm(hin, cv[L].Wk, cv[L].bk, k, NN_NODES, din, HC, 1, nullptr);
        gemm(hin, cv[L].Wv, cv[L].bv, v, NN_NODES, din, HC, 1, nullptr);
        gemm(hin, cv[L].Ws, cv[L].bs, r, NN_NODES, din, HC, 1, nullptr);
        // edge projection (K=16 zero-padded to BK=32)
        gemm(edge_attr, cv[L].We, nullptr, eproj, NE_EDGES, EDGE_DIM, HC, 0, nullptr);

        hipMemsetAsync(amaxU, 0, (size_t)NN_NODES * HEADS * sizeof(unsigned), stream);
        hipMemsetAsync(denom, 0, (size_t)NN_NODES * HEADS * sizeof(float), stream);
        hipMemsetAsync(outb,  0, NQ * sizeof(float), stream);

        int eh = NE_EDGES * HEADS;
        edge_alpha_kernel<<<(eh + 255) / 256, 256, 0, stream>>>(q, k, eproj, src, dst, alpha, amaxU);
        edge_exp_kernel<<<(eh + 255) / 256, 256, 0, stream>>>(alpha, amaxU, denom, dst);
        long long ec = (long long)NE_EDGES * HC;
        edge_scatter_kernel<<<(unsigned)((ec + 255) / 256), 256, 0, stream>>>(
            v, eproj, alpha, denom, src, dst, outb);
        beta_combine_kernel<<<(NN_NODES * 32 + 255) / 256, 256, 0, stream>>>(outb, r, cv[L].Wbeta);

        // transf: relu(outb @ W + b) then BN  (fused epilogue, mode 3)
        gemm(outb, tfW[L], tfB[L], h, NN_NODES, HC, EMB, 3, &bnp[L]);

        if (L >= 1)  // local = outputs of the 3 inner layers only
            add_into_kernel<<<(NN_NODES * EMB + 255) / 256, 256, 0, stream>>>(
                hsum, h, NN_NODES * EMB);

        hin = h;
        din = EMB;
    }

    // ---- GlobalAttention pooling ----
    hipMemsetAsync(gmaxU, 0, N_GRAPHS * sizeof(unsigned), stream);
    hipMemsetAsync(gden,  0, N_GRAPHS * sizeof(float), stream);
    hipMemsetAsync(pooled, 0, (size_t)N_GRAPHS * EMB * sizeof(float), stream);
    gate_kernel<<<(NN_NODES * 32 + 255) / 256, 256, 0, stream>>>(hsum, gateW, gateb, batch, gate, gmaxU);
    gate_exp_kernel<<<(NN_NODES + 255) / 256, 256, 0, stream>>>(gate, gmaxU, gden, batch);
    pool_scatter_kernel<<<(NN_NODES * EMB + 255) / 256, 256, 0, stream>>>(gate, gden, hsum, batch, pooled);

    // ---- MLP head ----
    gemm(pooled, l1W, l1b, z1, N_GRAPHS, EMB, DENSE, 2, nullptr);
    gemm(z1, l2W, l2b, z2, N_GRAPHS, DENSE, DENSE / 2, 2, nullptr);
    final_dot_kernel<<<(N_GRAPHS + 255) / 256, 256, 0, stream>>>(z2, l3W, l3b, (float*)d_out);
}
